// BasicTransformerBlock_47184510714017
// MI455X (gfx1250) — compile-verified, matrix operand order checked by reading, not run on hardware
//
#include <hip/hip_runtime.h>
#include <cstdint>

typedef __attribute__((ext_vector_type(16))) _Float16 v16h;
typedef __attribute__((ext_vector_type(8)))  _Float16 v8h;
typedef __attribute__((ext_vector_type(8)))  float    v8f;

#define NTOK   1024      // tokens per batch (self)
#define MCTX   256       // context tokens per batch
#define NB     8         // batch
#define DIMC   512       // model dim == heads*dhead == cdim
#define NHEAD  8
#define DHEAD  64
#define FFH    4096      // ff_w1 rows
#define FFI    2048      // GEGLU inner
#define LDST   40        // LDS row stride in halfs (80B: 16B aligned, conflict-free)

// ---------------------------------------------------------------- fragment load
__device__ __forceinline__ v16h frag_load(const _Float16* p0) {
  v8h lo = *(const v8h*)p0;
  v8h hi = *(const v8h*)(p0 + 16);
  v16h o;
#pragma unroll
  for (int i = 0; i < 8; ++i) { o[i] = lo[i]; o[i + 8] = hi[i]; }
  return o;
}
// A/B fragment from row-major storage (lane = row/col index, ISA 16-bit layout)
__device__ __forceinline__ v16h frag_ptr(const _Float16* base, int ld, int lane) {
  int r = lane & 15, kh = lane >> 4;
  return frag_load(base + (size_t)r * ld + kh * 8);
}

// ------------------------------------------------ CDNA5 async global->LDS copy
// Flat pointers to __shared__ carry the LDS byte offset in their low 32 bits
// (ISA 10.2, LDS aperture).
__device__ __forceinline__ unsigned lds_off(const void* p) {
  return (unsigned)(uintptr_t)p;
}
__device__ __forceinline__ void async_b128(unsigned lds, const _Float16* g) {
  asm volatile("global_load_async_to_lds_b128 %0, %1, off"
               :: "v"(lds), "v"(g) : "memory");
}
__device__ __forceinline__ void wait_async4() {
  asm volatile("s_wait_asynccnt 4" ::: "memory");
}
__device__ __forceinline__ void wait_async0() {
  asm volatile("s_wait_asynccnt 0" ::: "memory");
}

// ---------------------------------------------------------------- f32 -> f16
__global__ void k_cvt(const float* __restrict__ s, _Float16* __restrict__ d, int n) {
  int i = blockIdx.x * 256 + threadIdx.x;
  if (i < n) d[i] = (_Float16)s[i];
}

// ---------------------------------------------------------------- emb MLP (tiny)
__global__ void k_emb(const float* __restrict__ emb, const float* __restrict__ w,
                      const float* __restrict__ bias, float* __restrict__ out) {
  int b = blockIdx.x;
  __shared__ float se[128];
  if (threadIdx.x < 128) {
    float e = emb[b * 128 + threadIdx.x];
    se[threadIdx.x] = e / (1.0f + __expf(-e));
  }
  __syncthreads();
  for (int o = threadIdx.x; o < 2 * DIMC; o += 256) {
    float s = bias[o];
    const float* wr = w + (size_t)o * 128;
#pragma unroll 8
    for (int i = 0; i < 128; ++i) s += se[i] * wr[i];
    out[b * 2 * DIMC + o] = s;
  }
}

// ---------------------------------------------------------------- GroupNorm stats
__global__ void k_gnstats(const float* __restrict__ x, float* __restrict__ mu,
                          float* __restrict__ rs) {
  int bg = blockIdx.x;               // b*32+g
  int b = bg >> 5, g = bg & 31;
  const float* base = x + ((size_t)b * DIMC + g * 16) * NTOK;
  float s = 0.f, ss = 0.f;
  for (int i = threadIdx.x; i < 16 * NTOK; i += 256) {
    float v = base[i]; s += v; ss += v * v;
  }
#pragma unroll
  for (int m = 1; m < 32; m <<= 1) { s += __shfl_xor(s, m, 32); ss += __shfl_xor(ss, m, 32); }
  __shared__ float r1[8], r2[8];
  int wid = threadIdx.x >> 5, lane = threadIdx.x & 31;
  if (lane == 0) { r1[wid] = s; r2[wid] = ss; }
  __syncthreads();
  if (threadIdx.x == 0) {
    float S = 0.f, SS = 0.f;
    for (int i = 0; i < 8; ++i) { S += r1[i]; SS += r2[i]; }
    float m = S * (1.0f / 16384.0f);
    float var = SS * (1.0f / 16384.0f) - m * m;
    mu[bg] = m; rs[bg] = rsqrtf(var + 1e-6f);
  }
}

// -------------------------------------- GN + modulate + SiLU + transpose to token-major
__global__ void k_gnmod(const float* __restrict__ x, const float* __restrict__ embout,
                        const float* __restrict__ mu, const float* __restrict__ rs,
                        const float* __restrict__ gg, const float* __restrict__ gb,
                        _Float16* __restrict__ h16, float* __restrict__ xt) {
  __shared__ float tx[32][33];
  __shared__ float th[32][33];
  int b = blockIdx.z, c0 = blockIdx.y * 32, n0 = blockIdx.x * 32;
  int lx = threadIdx.x, ly = threadIdx.y;
#pragma unroll
  for (int k = 0; k < 4; ++k) {
    int cl = ly + k * 8, c = c0 + cl, g = c >> 4;
    float m = mu[b * 32 + g], rv = rs[b * 32 + g];
    float sc = 1.0f + embout[b * 2 * DIMC + c];
    float sh = embout[b * 2 * DIMC + DIMC + c];
    float xv = x[((size_t)b * DIMC + c) * NTOK + n0 + lx];
    float xn = (xv - m) * rv * gg[c] + gb[c];
    float v = xn * sc + sh;
    tx[cl][lx] = xv;
    th[cl][lx] = v / (1.0f + __expf(-v));
  }
  __syncthreads();
#pragma unroll
  for (int k = 0; k < 4; ++k) {
    int nl = ly + k * 8;
    size_t tok = (size_t)b * NTOK + n0 + nl;
    int c = c0 + lx;
    xt[tok * DIMC + c] = tx[lx][nl];
    h16[tok * DIMC + c] = (_Float16)th[lx][nl];
  }
}

// ---------------------------------------------------------------- LayerNorm (per token)
__global__ __launch_bounds__(128) void k_ln(const float* __restrict__ xt,
                                            const float* __restrict__ g,
                                            const float* __restrict__ bta,
                                            _Float16* __restrict__ y) {
  size_t t = blockIdx.x;
  const float* row = xt + t * DIMC;
  float vals[4], s = 0.f, ss = 0.f;
#pragma unroll
  for (int i = 0; i < 4; ++i) {
    vals[i] = row[threadIdx.x + i * 128];
    s += vals[i]; ss += vals[i] * vals[i];
  }
#pragma unroll
  for (int m = 1; m < 32; m <<= 1) { s += __shfl_xor(s, m, 32); ss += __shfl_xor(ss, m, 32); }
  __shared__ float a1[4], a2[4];
  int wid = threadIdx.x >> 5, lane = threadIdx.x & 31;
  if (lane == 0) { a1[wid] = s; a2[wid] = ss; }
  __syncthreads();
  s = a1[0] + a1[1] + a1[2] + a1[3];
  ss = a2[0] + a2[1] + a2[2] + a2[3];
  float mean = s * (1.0f / DIMC);
  float inv = rsqrtf(ss * (1.0f / DIMC) - mean * mean + 1e-5f);
#pragma unroll
  for (int i = 0; i < 4; ++i) {
    int c = threadIdx.x + i * 128;
    y[t * DIMC + c] = (_Float16)((vals[i] - mean) * inv * g[c] + bta[c]);
  }
}

// --------------------------- generic [b,C,N] f32 -> token-major f16 transpose (context)
__global__ void k_tr_cn(const float* __restrict__ src, _Float16* __restrict__ dst,
                        int C, int N) {
  __shared__ float t[32][33];
  int b = blockIdx.z, c0 = blockIdx.y * 32, n0 = blockIdx.x * 32;
  int lx = threadIdx.x, ly = threadIdx.y;
#pragma unroll
  for (int k = 0; k < 4; ++k)
    t[ly + k * 8][lx] = src[((size_t)b * C + c0 + ly + k * 8) * N + n0 + lx];
  __syncthreads();
#pragma unroll
  for (int k = 0; k < 4; ++k)
    dst[((size_t)b * N + n0 + ly + k * 8) * C + c0 + lx] = (_Float16)t[lx][ly + k * 8];
}

// --------------------------- token-major f32 -> [b,C,N] f32 (final output)
__global__ void k_tr_out(const float* __restrict__ xt, float* __restrict__ out) {
  __shared__ float t[32][33];
  int b = blockIdx.z, c0 = blockIdx.y * 32, n0 = blockIdx.x * 32;
  int lx = threadIdx.x, ly = threadIdx.y;
#pragma unroll
  for (int k = 0; k < 4; ++k)
    t[ly + k * 8][lx] = xt[((size_t)b * NTOK + n0 + ly + k * 8) * DIMC + c0 + lx];
  __syncthreads();
#pragma unroll
  for (int k = 0; k < 4; ++k)
    out[((size_t)b * DIMC + c0 + ly + k * 8) * NTOK + n0 + lx] = t[lx][ly + k * 8];
}

// --------------------------- V [tok,512] f16 -> Vt [bh][64][Nk] f16
__global__ void k_vt(const _Float16* __restrict__ V, _Float16* __restrict__ Vt, int Nk) {
  __shared__ _Float16 t[32][33];
  int bh = blockIdx.z, b = bh >> 3, h = bh & 7;
  int d0 = blockIdx.y * 32, n0 = blockIdx.x * 32;
  int lx = threadIdx.x, ly = threadIdx.y;
#pragma unroll
  for (int k = 0; k < 4; ++k)
    t[ly + k * 8][lx] = V[((size_t)b * Nk + n0 + ly + k * 8) * DIMC + h * DHEAD + d0 + lx];
  __syncthreads();
#pragma unroll
  for (int k = 0; k < 4; ++k)
    Vt[((size_t)bh * DHEAD + d0 + ly + k * 8) * Nk + n0 + lx] = t[lx][ly + k * 8];
}

// ---------------------------------------------------------------- GEGLU (exact gelu)
__global__ void k_geglu(const _Float16* __restrict__ ffh, _Float16* __restrict__ u) {
  size_t idx = (size_t)blockIdx.x * 256 + threadIdx.x;
  size_t t = idx / FFI; int j = (int)(idx % FFI);
  float a = (float)ffh[t * FFH + j];
  float g = (float)ffh[t * FFH + FFI + j];
  float ge = 0.5f * g * (1.0f + erff(g * 0.70710678118654752f));
  u[idx] = (_Float16)(a * ge);
}

// ================================================================ WMMA GEMM
// C[M,N] = A[M,K] (f16, row-major, lda) @ W[N,K]^T (f16, row-major, ldw)
// + bias[N] + resid[M,N];  f32 (OF) and/or f16 (OH) output, ldc.
// Block tile 128x128x32, 8 waves 2x4, wave tile 64x32 (8 wmma).
// Double-buffered LDS fed by GLOBAL_LOAD_ASYNC_TO_LDS_B128 (ASYNCcnt pipelined).
// M,N multiples of 128; K multiple of 64.
__device__ __forceinline__ void gemm_step(const _Float16* cA, const _Float16* cB,
                                          int wm, int wn, int r, int kh,
                                          v8f acc[4][2]) {
  v16h af[4], bf[2];
#pragma unroll
  for (int i = 0; i < 4; ++i)
    af[i] = frag_load(&cA[(wm * 64 + i * 16 + r) * LDST + kh * 8]);
#pragma unroll
  for (int j = 0; j < 2; ++j)
    bf[j] = frag_load(&cB[(wn * 32 + j * 16 + r) * LDST + kh * 8]);
#pragma unroll
  for (int i = 0; i < 4; ++i)
#pragma unroll
    for (int j = 0; j < 2; ++j)
      acc[i][j] = __builtin_amdgcn_wmma_f32_16x16x32_f16(
          false, af[i], false, bf[j], (short)0, acc[i][j], false, false);
}

template <bool BIAS, bool RESID, bool OF, bool OH>
__global__ __launch_bounds__(256) void k_gemm(
    const _Float16* __restrict__ A, int lda,
    const _Float16* __restrict__ W, int ldw,
    float* __restrict__ Cf, _Float16* __restrict__ Ch, int ldc,
    const float* __restrict__ bias, const float* __restrict__ resid, int K) {
  alignas(16) __shared__ _Float16 sA[2][128 * LDST];
  alignas(16) __shared__ _Float16 sB[2][128 * LDST];
  const int tid = threadIdx.x;
  const int lane = tid & 31, wid = tid >> 5;
  const int wm = wid >> 2, wn = wid & 3;
  const size_t bm = (size_t)blockIdx.x * 128, bn = (size_t)blockIdx.y * 128;

  const v8f vz = {0.f, 0.f, 0.f, 0.f, 0.f, 0.f, 0.f, 0.f};
  v8f acc[4][2];
#pragma unroll
  for (int i = 0; i < 4; ++i)
#pragma unroll
    for (int j = 0; j < 2; ++j) acc[i][j] = vz;

  // per-thread staging slot: 16 halfs of one row (2x b128)
  const int sRow = tid >> 1;
  const int sCol = (tid & 1) * 16;
  const _Float16* gA = A + (bm + sRow) * lda + sCol;
  const _Float16* gB = W + (bn + sRow) * ldw + sCol;
  const unsigned la[2] = { lds_off(&sA[0][sRow * LDST + sCol]),
                           lds_off(&sA[1][sRow * LDST + sCol]) };
  const unsigned lb[2] = { lds_off(&sB[0][sRow * LDST + sCol]),
                           lds_off(&sB[1][sRow * LDST + sCol]) };
  const int r = lane & 15, kh = lane >> 4;

  // prologue: tile k=0 -> buffer 0
  async_b128(la[0], gA);      async_b128(la[0] + 16, gA + 8);
  async_b128(lb[0], gB);      async_b128(lb[0] + 16, gB + 8);

  int k0 = 0;
  for (;;) {
    // ---- phase A: compute buf0 (tile k0), prefetch k0+32 -> buf1
    if (k0 + 32 < K) {
      const _Float16* nA = gA + 32; const _Float16* nB = gB + 32;
      async_b128(la[1], nA);  async_b128(la[1] + 16, nA + 8);
      async_b128(lb[1], nB);  async_b128(lb[1] + 16, nB + 8);
      wait_async4();
    } else {
      wait_async0();
    }
    __syncthreads();
    gemm_step(sA[0], sB[0], wm, wn, r, kh, acc);
    __syncthreads();
    if (k0 + 32 >= K) break;

    // ---- phase B: compute buf1 (tile k0+32), prefetch k0+64 -> buf0
    if (k0 + 64 < K) {
      const _Float16* nA = gA + 64; const _Float16* nB = gB + 64;
      async_b128(la[0], nA);  async_b128(la[0] + 16, nA + 8);
      async_b128(lb[0], nB);  async_b128(lb[0] + 16, nB + 8);
      wait_async4();
    } else {
      wait_async0();
    }
    __syncthreads();
    gemm_step(sA[1], sB[1], wm, wn, r, kh, acc);
    __syncthreads();
    if (k0 + 64 >= K) break;
    gA += 64; gB += 64; k0 += 64;
  }

  const int half = lane >> 4;
#pragma unroll
  for (int i = 0; i < 4; ++i)
#pragma unroll
    for (int j = 0; j < 2; ++j) {
      const size_t col = bn + wn * 32 + j * 16 + r;
      float bv = 0.0f;
      if (BIAS) bv = bias[col];
#pragma unroll
      for (int e = 0; e < 8; ++e) {
        const size_t row = bm + wm * 64 + i * 16 + (half ? e + 8 : e);
        float v = acc[i][j][e] + bv;
        if (RESID) v += resid[row * ldc + col];
        if (OF) Cf[row * ldc + col] = v;
        if (OH) Ch[row * ldc + col] = (_Float16)v;
      }
    }
}

// ================================================================ flash attention
// Per block: 4 waves, 64 Q rows of one (b,h). Wave owns 16 rows; online softmax.
// Q,K token-major [*,512]; Vt [bh][64][Nk]; O token-major f16 [*,512].
__global__ __launch_bounds__(128) void k_flash(
    const _Float16* __restrict__ Q, const _Float16* __restrict__ Km,
    const _Float16* __restrict__ Vt, _Float16* __restrict__ O,
    int Nk, int kTokPerB) {
  alignas(16) __shared__ _Float16 psh[4][16 * LDST];
  const int bh = blockIdx.y, b = bh >> 3, h = bh & 7;
  const int lane = threadIdx.x & 31, wid = threadIdx.x >> 5;
  const int r = lane & 15, half = lane >> 4;
  const int m0 = blockIdx.x * 64 + wid * 16;

  const _Float16* qb = Q + ((size_t)(b * NTOK + m0)) * DIMC + h * DHEAD;
  v16h qf0 = frag_ptr(qb, DIMC, lane);
  v16h qf1 = frag_ptr(qb + 32, DIMC, lane);

  const v8f vz = {0.f, 0.f, 0.f, 0.f, 0.f, 0.f, 0.f, 0.f};
  v8f accO[4];
#pragma unroll
  for (int t = 0; t < 4; ++t) accO[t] = vz;
  float mi[8], li[8];
#pragma unroll
  for (int e = 0; e < 8; ++e) { mi[e] = -1e30f; li[e] = 0.f; }

  const _Float16* kb0 = Km + ((size_t)b * kTokPerB) * DIMC + h * DHEAD;
  const _Float16* vtb = Vt + (size_t)bh * DHEAD * Nk;
  _Float16* pbuf = psh[wid];

  for (int j0 = 0; j0 < Nk; j0 += 32) {
    v8f s0 = vz, s1 = vz;
    {
      const _Float16* kb = kb0 + (size_t)j0 * DIMC;
      v16h kf0 = frag_ptr(kb, DIMC, lane);
      v16h kf1 = frag_ptr(kb + 32, DIMC, lane);
      s0 = __builtin_amdgcn_wmma_f32_16x16x32_f16(false, qf0, false, kf0, (short)0, s0, false, false);
      s0 = __builtin_amdgcn_wmma_f32_16x16x32_f16(false, qf1, false, kf1, (short)0, s0, false, false);
      const _Float16* kb2 = kb + (size_t)16 * DIMC;
      v16h kg0 = frag_ptr(kb2, DIMC, lane);
      v16h kg1 = frag_ptr(kb2 + 32, DIMC, lane);
      s1 = __builtin_amdgcn_wmma_f32_16x16x32_f16(false, qf0, false, kg0, (short)0, s1, false, false);
      s1 = __builtin_amdgcn_wmma_f32_16x16x32_f16(false, qf1, false, kg1, (short)0, s1, false, false);
    }
#pragma unroll
    for (int e = 0; e < 8; ++e) {
      float v0 = s0[e] * 0.125f, v1 = s1[e] * 0.125f;
      float mx = fmaxf(v0, v1);
#pragma unroll
      for (int m = 1; m < 16; m <<= 1) mx = fmaxf(mx, __shfl_xor(mx, m, 32));
      float mn = fmaxf(mi[e], mx);
      float alpha = __expf(mi[e] - mn);
      float p0 = __expf(v0 - mn), p1 = __expf(v1 - mn);
      float rs = p0 + p1;
#pragma unroll
      for (int m = 1; m < 16; m <<= 1) rs += __shfl_xor(rs, m, 32);
      li[e] = li[e] * alpha + rs;
      mi[e] = mn;
#pragma unroll
      for (int t = 0; t < 4; ++t) accO[t][e] *= alpha;
      int prow = half ? e + 8 : e;                   // C-layout row for this lane
      pbuf[prow * LDST + r] = (_Float16)p0;          // cols j0..j0+15
      pbuf[prow * LDST + 16 + r] = (_Float16)p1;     // cols j0+16..j0+31
    }
    __syncthreads();
    v16h pa = frag_ptr(pbuf, LDST, lane);            // P as A-fragment (16x32)
#pragma unroll
    for (int t = 0; t < 4; ++t) {
      v16h vf = frag_ptr(vtb + (size_t)(t * 16) * Nk + j0, Nk, lane);
      accO[t] = __builtin_amdgcn_wmma_f32_16x16x32_f16(
          false, pa, false, vf, (short)0, accO[t], false, false);
    }
    __syncthreads();
  }
  _Float16* ob = O + ((size_t)(b * NTOK + m0)) * DIMC + h * DHEAD;
#pragma unroll
  for (int t = 0; t < 4; ++t)
#pragma unroll
    for (int e = 0; e < 8; ++e) {
      int row = half ? e + 8 : e;
      ob[(size_t)row * DIMC + t * 16 + r] = (_Float16)(accO[t][e] / li[e]);
    }
}

// ================================================================ host
extern "C" void kernel_launch(void* const* d_in, const int* in_sizes, int n_in,
                              void* d_out, int out_size, void* d_ws, size_t ws_size,
                              hipStream_t stream) {
  (void)in_sizes; (void)n_in; (void)out_size; (void)ws_size;
  const float* x      = (const float*)d_in[0];
  const float* emb    = (const float*)d_in[1];
  const float* ctx    = (const float*)d_in[2];
  const float* w_emb  = (const float*)d_in[3];
  const float* b_emb  = (const float*)d_in[4];
  const float* gn_g   = (const float*)d_in[5];
  const float* gn_b   = (const float*)d_in[6];
  const float* conv_w = (const float*)d_in[7];
  const float* conv_b = (const float*)d_in[8];
  const float* a1_wq  = (const float*)d_in[9];
  const float* a1_wk  = (const float*)d_in[10];
  const float* a1_wv  = (const float*)d_in[11];
  const float* a1_wo  = (const float*)d_in[12];
  const float* a1_bo  = (const float*)d_in[13];
  const float* a2_wq  = (const float*)d_in[14];
  const float* a2_wk  = (const float*)d_in[15];
  const float* a2_wv  = (const float*)d_in[16];
  const float* a2_wo  = (const float*)d_in[17];
  const float* a2_bo  = (const float*)d_in[18];
  const float* ln1_g  = (const float*)d_in[19];
  const float* ln1_b  = (const float*)d_in[20];
  const float* ln2_g  = (const float*)d_in[21];
  const float* ln2_b  = (const float*)d_in[22];
  const float* ln3_g  = (const float*)d_in[23];
  const float* ln3_b  = (const float*)d_in[24];
  const float* ff_w1  = (const float*)d_in[25];
  const float* ff_b1  = (const float*)d_in[26];
  const float* ff_w2  = (const float*)d_in[27];
  const float* ff_b2  = (const float*)d_in[28];

  const int TOK = NB * NTOK;        // 8192
  const int CTOK = NB * MCTX;       // 2048

  char* ws = (char*)d_ws;
  size_t off = 0;
  auto alloc = [&](size_t bytes) -> void* {
    void* p = ws + off;
    off = (off + bytes + 255) & ~(size_t)255;
    return p;
  };
  _Float16* w16_conv = (_Float16*)alloc((size_t)DIMC * DIMC * 2);
  _Float16* w16_a1q  = (_Float16*)alloc((size_t)DIMC * DIMC * 2);
  _Float16* w16_a1k  = (_Float16*)alloc((size_t)DIMC * DIMC * 2);
  _Float16* w16_a1v  = (_Float16*)alloc((size_t)DIMC * DIMC * 2);
  _Float16* w16_a1o  = (_Float16*)alloc((size_t)DIMC * DIMC * 2);
  _Float16* w16_a2q  = (_Float16*)alloc((size_t)DIMC * DIMC * 2);
  _Float16* w16_a2k  = (_Float16*)alloc((size_t)DIMC * DIMC * 2);
  _Float16* w16_a2v  = (_Float16*)alloc((size_t)DIMC * DIMC * 2);
  _Float16* w16_a2o  = (_Float16*)alloc((size_t)DIMC * DIMC * 2);
  _Float16* w16_ff1  = (_Float16*)alloc((size_t)FFH * DIMC * 2);
  _Float16* w16_ff2  = (_Float16*)alloc((size_t)DIMC * FFI * 2);
  float*    embout   = (float*)alloc((size_t)NB * 2 * DIMC * 4);
  float*    mu       = (float*)alloc(256 * 4);
  float*    rsg      = (float*)alloc(256 * 4);
  float*    xt       = (float*)alloc((size_t)TOK * DIMC * 4);      // residual stream
  _Float16* h16      = (_Float16*)alloc((size_t)TOK * DIMC * 2);   // act / attn-out
  _Float16* y16      = (_Float16*)alloc((size_t)TOK * DIMC * 2);   // ln output
  _Float16* q16      = (_Float16*)alloc((size_t)TOK * DIMC * 2);
  _Float16* k16      = (_Float16*)alloc((size_t)TOK * DIMC * 2);
  _Float16* v16      = (_Float16*)alloc((size_t)TOK * DIMC * 2);
  _Float16* vt16     = (_Float16*)alloc((size_t)NB * NHEAD * DHEAD * NTOK * 2);
  _Float16* ctx16    = (_Float16*)alloc((size_t)CTOK * DIMC * 2);
  _Float16* ffh16    = (_Float16*)alloc((size_t)TOK * FFH * 2);
  _Float16* u16      = (_Float16*)alloc((size_t)TOK * FFI * 2);

  auto cvt = [&](const float* s, _Float16* d, int n) {
    k_cvt<<<dim3((n + 255) / 256), dim3(256), 0, stream>>>(s, d, n);
  };
  cvt(conv_w, w16_conv, DIMC * DIMC);
  cvt(a1_wq, w16_a1q, DIMC * DIMC);  cvt(a1_wk, w16_a1k, DIMC * DIMC);
  cvt(a1_wv, w16_a1v, DIMC * DIMC);  cvt(a1_wo, w16_a1o, DIMC * DIMC);
  cvt(a2_wq, w16_a2q, DIMC * DIMC);  cvt(a2_wk, w16_a2k, DIMC * DIMC);
  cvt(a2_wv, w16_a2v, DIMC * DIMC);  cvt(a2_wo, w16_a2o, DIMC * DIMC);
  cvt(ff_w1, w16_ff1, FFH * DIMC);   cvt(ff_w2, w16_ff2, DIMC * FFI);

  // bias + residual -> f32 out
  auto gemmBR = [&](const _Float16* A, int lda, const _Float16* W, int ldw,
                    float* Cf, int ldc, const float* bias, const float* resid,
                    int M, int N, int K) {
    k_gemm<true, true, true, false><<<dim3(M / 128, N / 128), dim3(256), 0, stream>>>(
        A, lda, W, ldw, Cf, nullptr, ldc, bias, resid, K);
  };
  // plain -> f16 out
  auto gemmH = [&](const _Float16* A, int lda, const _Float16* W, int ldw,
                   _Float16* Ch, int ldc, int M, int N, int K) {
    k_gemm<false, false, false, true><<<dim3(M / 128, N / 128), dim3(256), 0, stream>>>(
        A, lda, W, ldw, nullptr, Ch, ldc, nullptr, nullptr, K);
  };
  // bias -> f16 out
  auto gemmHB = [&](const _Float16* A, int lda, const _Float16* W, int ldw,
                    _Float16* Ch, int ldc, const float* bias, int M, int N, int K) {
    k_gemm<true, false, false, true><<<dim3(M / 128, N / 128), dim3(256), 0, stream>>>(
        A, lda, W, ldw, nullptr, Ch, ldc, bias, nullptr, K);
  };

  // ---- residual conv branch
  k_emb<<<dim3(NB), dim3(256), 0, stream>>>(emb, w_emb, b_emb, embout);
  k_gnstats<<<dim3(NB * 32), dim3(256), 0, stream>>>(x, mu, rsg);
  k_gnmod<<<dim3(NTOK / 32, DIMC / 32, NB), dim3(32, 8), 0, stream>>>(
      x, embout, mu, rsg, gn_g, gn_b, h16, xt);
  gemmBR(h16, DIMC, w16_conv, DIMC, xt, DIMC, conv_b, xt, TOK, DIMC, DIMC);

  // ---- self attention
  k_ln<<<dim3(TOK), dim3(128), 0, stream>>>(xt, ln1_g, ln1_b, y16);
  gemmH(y16, DIMC, w16_a1q, DIMC, q16, DIMC, TOK, DIMC, DIMC);
  gemmH(y16, DIMC, w16_a1k, DIMC, k16, DIMC, TOK, DIMC, DIMC);
  gemmH(y16, DIMC, w16_a1v, DIMC, v16, DIMC, TOK, DIMC, DIMC);
  k_vt<<<dim3(NTOK / 32, 2, NB * NHEAD), dim3(32, 8), 0, stream>>>(v16, vt16, NTOK);
  k_flash<<<dim3(NTOK / 64, NB * NHEAD), dim3(128), 0, stream>>>(
      q16, k16, vt16, h16, NTOK, NTOK);
  gemmBR(h16, DIMC, w16_a1o, DIMC, xt, DIMC, a1_bo, xt, TOK, DIMC, DIMC);

  // ---- cross attention
  k_ln<<<dim3(TOK), dim3(128), 0, stream>>>(xt, ln2_g, ln2_b, y16);
  gemmH(y16, DIMC, w16_a2q, DIMC, q16, DIMC, TOK, DIMC, DIMC);
  k_tr_cn<<<dim3(MCTX / 32, DIMC / 32, NB), dim3(32, 8), 0, stream>>>(ctx, ctx16, DIMC, MCTX);
  gemmH(ctx16, DIMC, w16_a2k, DIMC, k16, DIMC, CTOK, DIMC, DIMC);
  gemmH(ctx16, DIMC, w16_a2v, DIMC, v16, DIMC, CTOK, DIMC, DIMC);
  k_vt<<<dim3(MCTX / 32, 2, NB * NHEAD), dim3(32, 8), 0, stream>>>(v16, vt16, MCTX);
  k_flash<<<dim3(NTOK / 64, NB * NHEAD), dim3(128), 0, stream>>>(
      q16, k16, vt16, h16, MCTX, MCTX);
  gemmBR(h16, DIMC, w16_a2o, DIMC, xt, DIMC, a2_bo, xt, TOK, DIMC, DIMC);

  // ---- GEGLU feed-forward
  k_ln<<<dim3(TOK), dim3(128), 0, stream>>>(xt, ln3_g, ln3_b, y16);
  gemmHB(y16, DIMC, w16_ff1, DIMC, ffh16, FFH, ff_b1, TOK, FFH, DIMC);
  k_geglu<<<dim3((unsigned)((size_t)TOK * FFI / 256)), dim3(256), 0, stream>>>(ffh16, u16);
  gemmBR(u16, FFI, w16_ff2, FFI, xt, DIMC, ff_b2, xt, TOK, DIMC, FFI);

  // ---- back to [B, dim, N]
  k_tr_out<<<dim3(NTOK / 32, DIMC / 32, NB), dim3(32, 8), 0, stream>>>(xt, (float*)d_out);
}
